// AttentionDecoderL_19679540150694
// MI455X (gfx1250) — matrix operand order repaired and checked by reading, not compiled, not run on hardware
//
#include <hip/hip_runtime.h>
#include <hip/hip_bf16.h>

#define HID   768
#define CTX   384
#define NH    10
#define BATCH 32
#define SEQ   2048
#define ROWS  (BATCH * SEQ)       // 65536
#define NT    3
#define NO    64

typedef __attribute__((ext_vector_type(16))) __bf16 v16bf;
typedef __attribute__((ext_vector_type(8)))  float  v8f;

union AB { v16bf v; uint4 q[2]; };

__device__ __forceinline__ unsigned short f2bf(float f) {
    union { float f; unsigned u; } x; x.f = f;
    unsigned r = x.u + 0x7FFFu + ((x.u >> 16) & 1u);   // RNE
    return (unsigned short)(r >> 16);
}
__device__ __forceinline__ float bf2f(unsigned short b) {
    union { unsigned u; float f; } x; x.u = ((unsigned)b) << 16; return x.f;
}

// ---------------------------------------------------------------- weight cvt
__global__ void k_cvt(const float* __restrict__ s, unsigned short* __restrict__ d, int n) {
    int i = blockIdx.x * 256 + threadIdx.x;
    if (i < n) d[i] = f2bf(s[i]);
}

__global__ void k_zero(float* __restrict__ p, int n) {
    int i = blockIdx.x * 256 + threadIdx.x;
    if (i < n) p[i] = 0.0f;
}

// ---------------------------------------------------------------- fused per-row pipeline
// grid: ROWS/32 = 2048 workgroups, 256 threads (8 waves). Each WG owns 32 token rows.
// Phase 0: stage x[32 x 768] as bf16 in LDS.
// Phase 1: h = swish(x @ Wff^T + b_ff), ScaleNorm -> overwrite LDS with h_norm bf16,
//          also spill h_norm bf16 to global for the pooling pass.
// Phase 2: p = tanh(h_norm @ Wpool^T + b_pool) -> LDS f32.
// Phase 3: scores[hd, row] = p . context[hd].
__global__ __launch_bounds__(256)
void k_fused_rows(const float* __restrict__ x,
                  const unsigned short* __restrict__ Wffb,    // [768][768] bf16
                  const float* __restrict__ b_ff,
                  const unsigned short* __restrict__ Wpoolb,  // [384][768] bf16
                  const float* __restrict__ b_pool,
                  const float* __restrict__ context,          // [10][384] f32
                  const float* __restrict__ gptr,
                  unsigned short* __restrict__ h_out,         // [ROWS][768] bf16
                  float* __restrict__ scores)                 // [10][ROWS] f32
{
    __shared__ unsigned short lds_x[32 * HID];   // 49152 B (x, then h_norm)
    __shared__ float          lds_p[32 * CTX];   // 49152 B
    __shared__ float          lds_ss[32];        // row sum-of-squares

    const int tid  = threadIdx.x;
    const int r0   = blockIdx.x * 32;
    const int wave = tid >> 5;
    const int lane = tid & 31;
    const int ln   = lane & 15;
    const int hi   = lane >> 4;

    // ---- Phase 0: cooperative load x block -> bf16 LDS (coalesced float4)
    {
        const float4* xs = (const float4*)(x + (size_t)r0 * HID);
        for (int i = tid; i < 32 * HID / 4; i += 256) {
            float4 f = xs[i];
            unsigned lo = (unsigned)f2bf(f.x) | ((unsigned)f2bf(f.y) << 16);
            unsigned hiw = (unsigned)f2bf(f.z) | ((unsigned)f2bf(f.w) << 16);
            ((uint2*)lds_x)[i] = make_uint2(lo, hiw);
        }
    }
    __syncthreads();

    // ---- Phase 1: ff GEMM, 32 rows x 96 cols per wave (12 accum tiles)
    const int n0 = wave * 96;
    v8f acc[2][6];
    #pragma unroll
    for (int i = 0; i < 2; ++i)
        #pragma unroll
        for (int t = 0; t < 6; ++t)
            #pragma unroll
            for (int r = 0; r < 8; ++r) acc[i][t][r] = 0.0f;

    for (int k0 = 0; k0 < HID; k0 += 32) {
        AB a0, a1;
        const int kb = k0 + hi * 8;          // ISA A layout: lo lanes k0..7/16..23, hi lanes +8
        a0.q[0] = *(const uint4*)&lds_x[ln * HID + kb];
        a0.q[1] = *(const uint4*)&lds_x[ln * HID + kb + 16];
        a1.q[0] = *(const uint4*)&lds_x[(ln + 16) * HID + kb];
        a1.q[1] = *(const uint4*)&lds_x[(ln + 16) * HID + kb + 16];
        const int kbb = k0 + hi * 8;         // B layout matches A: lo lanes k0..7/16..23, hi lanes +8
        #pragma unroll
        for (int t = 0; t < 6; ++t) {
            AB b;
            const unsigned short* wp = Wffb + (size_t)(n0 + t * 16 + ln) * HID + kbb;
            b.q[0] = ((const uint4*)wp)[0];
            b.q[1] = ((const uint4*)wp)[2];
            __builtin_prefetch((const void*)(wp + 32), 0, 1);
            acc[0][t] = __builtin_amdgcn_wmma_f32_16x16x32_bf16(false, a0.v, false, b.v,
                                                                (short)0, acc[0][t], false, false);
            acc[1][t] = __builtin_amdgcn_wmma_f32_16x16x32_bf16(false, a1.v, false, b.v,
                                                                (short)0, acc[1][t], false, false);
        }
    }

    // ---- epilogue: bias + swish + per-row sum of squares
    float rowss[16];
    #pragma unroll
    for (int i = 0; i < 16; ++i) rowss[i] = 0.0f;
    #pragma unroll
    for (int t = 0; t < 6; ++t) {
        float bv = b_ff[n0 + t * 16 + ln];
        #pragma unroll
        for (int r = 0; r < 8; ++r) {
            float v = acc[0][t][r] + bv;
            v = v / (1.0f + __expf(-v));            // swish
            acc[0][t][r] = v;
            rowss[r] += v * v;
            float v1 = acc[1][t][r] + bv;
            v1 = v1 / (1.0f + __expf(-v1));
            acc[1][t][r] = v1;
            rowss[8 + r] += v1 * v1;
        }
    }
    // reduce over the 16 lanes of each half-wave (columns of this strip)
    #pragma unroll
    for (int m = 1; m < 16; m <<= 1)
        #pragma unroll
        for (int i = 0; i < 16; ++i) rowss[i] += __shfl_xor(rowss[i], m, 32);

    if (tid < 32) lds_ss[tid] = 0.0f;
    __syncthreads();
    if (ln == 0) {   // lane 0 and lane 16 of each wave carry half-wave totals
        #pragma unroll
        for (int r = 0; r < 8; ++r) {
            atomicAdd(&lds_ss[hi * 8 + r],       rowss[r]);
            atomicAdd(&lds_ss[16 + hi * 8 + r],  rowss[8 + r]);
        }
    }
    __syncthreads();

    const float g = gptr[0];
    float scl[16];
    #pragma unroll
    for (int r = 0; r < 8; ++r) {
        scl[r]     = g / fmaxf(sqrtf(lds_ss[hi * 8 + r]),      1e-6f);
        scl[8 + r] = g / fmaxf(sqrtf(lds_ss[16 + hi * 8 + r]), 1e-6f);
    }

    // scale, overwrite LDS with h_norm bf16, spill bf16 h to global
    #pragma unroll
    for (int t = 0; t < 6; ++t) {
        int col = n0 + t * 16 + ln;
        #pragma unroll
        for (int r = 0; r < 8; ++r) {
            int m0 = hi * 8 + r;
            unsigned short b0 = f2bf(acc[0][t][r] * scl[r]);
            lds_x[m0 * HID + col] = b0;
            h_out[(size_t)(r0 + m0) * HID + col] = b0;
            int m1 = 16 + hi * 8 + r;
            unsigned short b1 = f2bf(acc[1][t][r] * scl[8 + r]);
            lds_x[m1 * HID + col] = b1;
            h_out[(size_t)(r0 + m1) * HID + col] = b1;
        }
    }
    __syncthreads();

    // ---- Phase 2: pool GEMM, 32 rows x 48 cols per wave (6 accum tiles)
    const int c0 = wave * 48;
    v8f accp[2][3];
    #pragma unroll
    for (int i = 0; i < 2; ++i)
        #pragma unroll
        for (int t = 0; t < 3; ++t)
            #pragma unroll
            for (int r = 0; r < 8; ++r) accp[i][t][r] = 0.0f;

    for (int k0 = 0; k0 < HID; k0 += 32) {
        AB a0, a1;
        const int kb = k0 + hi * 8;
        a0.q[0] = *(const uint4*)&lds_x[ln * HID + kb];
        a0.q[1] = *(const uint4*)&lds_x[ln * HID + kb + 16];
        a1.q[0] = *(const uint4*)&lds_x[(ln + 16) * HID + kb];
        a1.q[1] = *(const uint4*)&lds_x[(ln + 16) * HID + kb + 16];
        const int kbb = k0 + hi * 8;
        #pragma unroll
        for (int t = 0; t < 3; ++t) {
            AB b;
            const unsigned short* wp = Wpoolb + (size_t)(c0 + t * 16 + ln) * HID + kbb;
            b.q[0] = ((const uint4*)wp)[0];
            b.q[1] = ((const uint4*)wp)[2];
            accp[0][t] = __builtin_amdgcn_wmma_f32_16x16x32_bf16(false, a0.v, false, b.v,
                                                                 (short)0, accp[0][t], false, false);
            accp[1][t] = __builtin_amdgcn_wmma_f32_16x16x32_bf16(false, a1.v, false, b.v,
                                                                 (short)0, accp[1][t], false, false);
        }
    }
    #pragma unroll
    for (int t = 0; t < 3; ++t) {
        int col = c0 + t * 16 + ln;
        float bv = b_pool[col];
        #pragma unroll
        for (int r = 0; r < 8; ++r) {
            lds_p[(hi * 8 + r) * CTX + col]      = tanhf(accp[0][t][r] + bv);
            lds_p[(16 + hi * 8 + r) * CTX + col] = tanhf(accp[1][t][r] + bv);
        }
    }
    __syncthreads();

    // ---- Phase 3: scores[hd][row] = p[row] . context[hd]
    for (int i = tid; i < 32 * NH; i += 256) {
        int row = i / NH, hd = i % NH;
        const float4* pp = (const float4*)&lds_p[row * CTX];
        const float4* cc = (const float4*)(context + (size_t)hd * CTX);
        float s = 0.0f;
        #pragma unroll 4
        for (int c = 0; c < CTX / 4; ++c) {
            float4 a = pp[c], b = cc[c];
            s += a.x * b.x + a.y * b.y + a.z * b.z + a.w * b.w;
        }
        scores[(size_t)hd * ROWS + (r0 + row)] = s;
    }
}

// ---------------------------------------------------------------- sigsoftmax over S
// grid: NH*BATCH = 320 workgroups, 256 threads each
__global__ __launch_bounds__(256)
void k_sigsoftmax(const float* __restrict__ scores,
                  const unsigned char* __restrict__ mask,
                  float* __restrict__ attn)
{
    const int hd = blockIdx.x / BATCH;
    const int b  = blockIdx.x % BATCH;
    const float* srow = scores + (size_t)hd * ROWS + (size_t)b * SEQ;
    const unsigned char* mrow = mask + (size_t)b * SEQ;
    float* arow = attn + (size_t)hd * ROWS + (size_t)b * SEQ;

    __shared__ float smax[8], ssum[8];
    const int tid = threadIdx.x;

    float mx = -3.4e38f;
    for (int s = tid; s < SEQ; s += 256) mx = fmaxf(mx, srow[s]);   // max is unmasked (matches ref)
    #pragma unroll
    for (int m = 16; m >= 1; m >>= 1) mx = fmaxf(mx, __shfl_xor(mx, m, 32));
    if ((tid & 31) == 0) smax[tid >> 5] = mx;
    __syncthreads();
    float M = smax[0];
    #pragma unroll
    for (int i = 1; i < 8; ++i) M = fmaxf(M, smax[i]);

    float sum = 0.0f;
    for (int s = tid; s < SEQ; s += 256) {
        float v = srow[s];
        float e = __expf(v - M) / (1.0f + __expf(-v));  // exp(v-M)*sigmoid(v)
        if (!mrow[s]) e = 0.0f;
        arow[s] = e;
        sum += e;
    }
    #pragma unroll
    for (int m = 16; m >= 1; m >>= 1) sum += __shfl_xor(sum, m, 32);
    if ((tid & 31) == 0) ssum[tid >> 5] = sum;
    __syncthreads();
    float tot = 0.0f;
    #pragma unroll
    for (int i = 0; i < 8; ++i) tot += ssum[i];
    float inv = 1.0f / (tot + 1e-12f);
    for (int s = tid; s < SEQ; s += 256) arow[s] *= inv;
}

// ---------------------------------------------------------------- weighted pooling
// pooled[b, hd, d] = sum_s attn[hd,b,s] * h[b,s,d]; grid: 32 batches x 8 s-chunks
__global__ __launch_bounds__(256)
void k_pool(const float* __restrict__ attn,
            const unsigned short* __restrict__ hbf,
            float* __restrict__ pooled)
{
    const int b  = blockIdx.x >> 3;
    const int s0 = (blockIdx.x & 7) * 256;
    const int tid = threadIdx.x;
    __shared__ float la[NH * 256];
    for (int i = tid; i < NH * 256; i += 256) {
        int hd = i >> 8, sl = i & 255;
        la[i] = attn[(size_t)hd * ROWS + (size_t)b * SEQ + s0 + sl];
    }
    __syncthreads();

    float acc[3][NH];
    #pragma unroll
    for (int j = 0; j < 3; ++j)
        #pragma unroll
        for (int hd = 0; hd < NH; ++hd) acc[j][hd] = 0.0f;

    for (int sl = 0; sl < 256; ++sl) {
        const unsigned short* hp = hbf + (size_t)((size_t)b * SEQ + s0 + sl) * HID;
        float h0 = bf2f(hp[tid]);
        float h1 = bf2f(hp[tid + 256]);
        float h2 = bf2f(hp[tid + 512]);
        #pragma unroll
        for (int hd = 0; hd < NH; ++hd) {
            float a = la[hd * 256 + sl];
            acc[0][hd] += a * h0;
            acc[1][hd] += a * h1;
            acc[2][hd] += a * h2;
        }
    }
    #pragma unroll
    for (int j = 0; j < 3; ++j) {
        int d = tid + j * 256;
        #pragma unroll
        for (int hd = 0; hd < NH; ++hd)
            atomicAdd(&pooled[(size_t)b * (NH * HID) + (size_t)hd * HID + d], acc[j][hd]);
    }
}

// ---------------------------------------------------------------- logits = pooled @ W_post^T + b_post
// grid: 768/2 = 384 workgroups; each stages 2 rows of W_post in LDS
__global__ __launch_bounds__(256)
void k_post(const float* __restrict__ pooled, const float* __restrict__ W_post,
            const float* __restrict__ b_post, float* __restrict__ logits)
{
    __shared__ float lw[2 * NH * HID];   // 61440 B
    const int j0 = blockIdx.x * 2;
    const int tid = threadIdx.x;
    const int KP = NH * HID;             // 7680
    for (int i = tid; i < 2 * KP; i += 256) lw[i] = W_post[(size_t)j0 * KP + i];
    __syncthreads();
    const int b = tid >> 3, sub = tid & 7;
    #pragma unroll
    for (int jj = 0; jj < 2; ++jj) {
        const float* pr = pooled + (size_t)b * KP;
        const float* wr = lw + jj * KP;
        float acc = 0.0f;
        for (int k = sub; k < KP; k += 8) acc += pr[k] * wr[k];
        #pragma unroll
        for (int m = 1; m < 8; m <<= 1) acc += __shfl_xor(acc, m, 32);
        if (sub == 0) logits[(size_t)b * HID + j0 + jj] = acc + b_post[j0 + jj];
    }
}

// ---------------------------------------------------------------- out = logits @ W_out^T + b_out
// grid: 192 workgroups (one output column each)
__global__ __launch_bounds__(256)
void k_out(const float* __restrict__ logits, const float* __restrict__ W_out,
           const float* __restrict__ b_out, float* __restrict__ out)
{
    __shared__ float lw[HID];
    const int j = blockIdx.x;
    const int tid = threadIdx.x;
    for (int i = tid; i < HID; i += 256) lw[i] = W_out[(size_t)j * HID + i];
    __syncthreads();
    const int b = tid >> 3, sub = tid & 7;
    float acc = 0.0f;
    for (int k = sub; k < HID; k += 8) acc += logits[(size_t)b * HID + k] * lw[k];
    #pragma unroll
    for (int m = 1; m < 8; m <<= 1) acc += __shfl_xor(acc, m, 32);
    if (sub == 0) out[(size_t)b * (NT * NO) + j] = acc + b_out[j];
}

// ----------------------------------------------------------------
extern "C" void kernel_launch(void* const* d_in, const int* in_sizes, int n_in,
                              void* d_out, int out_size, void* d_ws, size_t ws_size,
                              hipStream_t stream) {
    (void)in_sizes; (void)n_in; (void)out_size; (void)ws_size;
    const float*         x       = (const float*)d_in[0];
    const unsigned char* mask    = (const unsigned char*)d_in[1];
    const float*         W_ff    = (const float*)d_in[2];
    const float*         b_ff    = (const float*)d_in[3];
    const float*         W_pool  = (const float*)d_in[4];
    const float*         b_pool  = (const float*)d_in[5];
    const float*         context = (const float*)d_in[6];
    const float*         W_post  = (const float*)d_in[7];
    const float*         b_post  = (const float*)d_in[8];
    const float*         W_out   = (const float*)d_in[9];
    const float*         b_out   = (const float*)d_in[10];
    const float*         g       = (const float*)d_in[11];
    float* out = (float*)d_out;

    char* ws = (char*)d_ws;
    const size_t OFF_WFF   = 0;
    const size_t OFF_WPOOL = OFF_WFF   + (size_t)HID * HID * 2;        // 1,179,648
    const size_t OFF_H     = OFF_WPOOL + (size_t)CTX * HID * 2;        // +589,824
    const size_t OFF_SC    = OFF_H     + (size_t)ROWS * HID * 2;       // +100,663,296
    const size_t OFF_AT    = OFF_SC    + (size_t)NH * ROWS * 4;        // +2,621,440
    const size_t OFF_PL    = OFF_AT    + (size_t)NH * ROWS * 4;
    const size_t OFF_LG    = OFF_PL    + (size_t)BATCH * NH * HID * 4;

    unsigned short* Wffb   = (unsigned short*)(ws + OFF_WFF);
    unsigned short* Wpoolb = (unsigned short*)(ws + OFF_WPOOL);
    unsigned short* hbf    = (unsigned short*)(ws + OFF_H);
    float* scores = (float*)(ws + OFF_SC);
    float* attn   = (float*)(ws + OFF_AT);
    float* pooled = (float*)(ws + OFF_PL);
    float* logits = (float*)(ws + OFF_LG);

    k_cvt<<<(HID * HID) / 256, 256, 0, stream>>>(W_ff, Wffb, HID * HID);
    k_cvt<<<(CTX * HID) / 256, 256, 0, stream>>>(W_pool, Wpoolb, CTX * HID);
    k_zero<<<(BATCH * NH * HID) / 256, 256, 0, stream>>>(pooled, BATCH * NH * HID);

    k_fused_rows<<<ROWS / 32, 256, 0, stream>>>(x, Wffb, b_ff, Wpoolb, b_pool,
                                                context, g, hbf, scores);
    k_sigsoftmax<<<NH * BATCH, 256, 0, stream>>>(scores, mask, attn);
    k_pool<<<BATCH * 8, 256, 0, stream>>>(attn, hbf, pooled);
    k_post<<<HID / 2, 256, 0, stream>>>(pooled, W_post, b_post, logits);
    k_out<<<NT * NO, 256, 0, stream>>>(logits, W_out, b_out, out);
}
